// DenseGINConv_75239237091643
// MI455X (gfx1250) — compile-verified, matrix operand order checked by reading, not run on hardware
//
#include <hip/hip_runtime.h>

typedef __attribute__((ext_vector_type(16))) _Float16 v16h;
typedef __attribute__((ext_vector_type(8)))  float    v8f;
typedef __attribute__((ext_vector_type(4)))  float    v4f;

#define NC 128   // channels
#define JT 8     // 16-wide output column tiles per row (128/16)
#define KC 4     // k-chunks of 32 (128/32)

// ---------------- h = (1+eps)*x  (vectorized stream) ----------------
__global__ void gin_init(const float4* __restrict__ x, const float* __restrict__ eps,
                         float4* __restrict__ h, int total4) {
  int i = blockIdx.x * blockDim.x + threadIdx.x;
  if (i >= total4) return;
  float s = 1.0f + eps[0];
  float4 v = x[i];
  v.x *= s; v.y *= s; v.z *= s; v.w *= s;
  h[i] = v;
}

// ---------------- scatter-add: h[dst] += x[src] ----------------
// one wave per edge; lane handles a 4-float slice of the 128-float row.
__global__ void gin_scatter(const float* __restrict__ x, const int* __restrict__ src,
                            const int* __restrict__ dst, float* __restrict__ h, int E_) {
  long long t = (long long)blockIdx.x * blockDim.x + threadIdx.x;
  int e    = (int)(t >> 5);
  int part = (int)(t & 31);
  if (e >= E_) return;
  int s = src[e];
  int d = dst[e];
  const float4 v = *(const float4*)(x + (size_t)s * NC + part * 4);
  float* p = h + (size_t)d * NC + part * 4;
  unsafeAtomicAdd(p + 0, v.x);
  unsafeAtomicAdd(p + 1, v.y);
  unsafeAtomicAdd(p + 2, v.z);
  unsafeAtomicAdd(p + 3, v.w);
}

// ---------------- pack W1/W2 into WMMA B-operand fragments ----------------
// frags layout: [layer][jtile][chunk][lane][16 halves]
// B operand (32x16, f16): lane L holds column j = jtile*16 + (L&15);
//   lanes 0-15:  K = chunk*32 + {0..7, 16..23}
//   lanes 16-31: K = chunk*32 + {8..15, 24..31}
__global__ void gin_pack_w(const float* __restrict__ W1, const float* __restrict__ W2,
                           _Float16* __restrict__ frags) {
  for (int entry = threadIdx.x; entry < 2 * JT * KC * 32; entry += blockDim.x) {
    int lane  = entry & 31;
    int chunk = (entry >> 5) & 3;
    int jt    = (entry >> 7) & 7;
    int layer = entry >> 10;
    const float* W = layer ? W2 : W1;   // stored [in(k), out(j)] row-major
    int j     = jt * 16 + (lane & 15);
    int kbase = chunk * 32 + ((lane & 16) ? 8 : 0);
    _Float16* o = frags + (size_t)entry * 16;
#pragma unroll
    for (int e = 0; e < 16; ++e) {
      int k = kbase + e + ((e >= 8) ? 8 : 0);
      o[e] = (_Float16)W[k * NC + j];
    }
  }
}

// ---------------- fused GEMM: out = relu(in @ W + bvec) [+bias2, relu] ------
// 8 waves/block, each wave owns a 16-row tile; in-place (in==out) is safe
// because all A fragments are register-resident before any store.
__global__ void __launch_bounds__(256)
gin_gemm(const float* __restrict__ in, float* __restrict__ out,
         const _Float16* __restrict__ frags, const float* __restrict__ bvec,
         const float* __restrict__ bias2, int n) {
  const int lane = threadIdx.x & 31;
  const int wave = threadIdx.x >> 5;
  const int tile = blockIdx.x * 8 + wave;
  const int m0   = tile * 16;
  if (m0 >= n) return;                       // wave-uniform
  int row  = m0 + (lane & 15);
  int rowc = (row < n) ? row : (n - 1);      // clamp tail loads (stores masked)
  const float* arow = in + (size_t)rowc * NC;
  const int kbase = (lane & 16) ? 8 : 0;

  // A operand (16x32 f16 per chunk): lane&15 = row, K split by lane half.
  v16h A[KC];
#pragma unroll
  for (int c = 0; c < KC; ++c) {
    int k0 = c * 32 + kbase;
    v4f f0 = *(const v4f*)(arow + k0);
    v4f f1 = *(const v4f*)(arow + k0 + 4);
    v4f f2 = *(const v4f*)(arow + k0 + 16);
    v4f f3 = *(const v4f*)(arow + k0 + 20);
    v16h a;
#pragma unroll
    for (int e = 0; e < 4; ++e) {
      a[e]      = (_Float16)f0[e];
      a[4 + e]  = (_Float16)f1[e];
      a[8 + e]  = (_Float16)f2[e];
      a[12 + e] = (_Float16)f3[e];
    }
    A[c] = a;
  }

#pragma unroll
  for (int jt = 0; jt < JT; ++jt) {
    int j = jt * 16 + (lane & 15);
    float bj = bvec[j];
    v8f acc = {bj, bj, bj, bj, bj, bj, bj, bj};
#pragma unroll
    for (int ch = 0; ch < KC; ++ch) {
      const v16h b = *(const v16h*)(frags + (((size_t)jt * KC + ch) * 32 + lane) * 16);
      acc = __builtin_amdgcn_wmma_f32_16x16x32_f16(
          /*neg_a=*/false, A[ch], /*neg_b=*/false, b,
          /*c_mod=*/(short)0, acc, /*reuse_a=*/false, /*reuse_b=*/false);
    }
    // epilogue: relu (+ final bias + relu for layer 2), write D tile
    // D layout: VGPR r -> row m0 + (lane<16 ? r : r+8), col = j
#pragma unroll
    for (int r = 0; r < 8; ++r) {
      int rr = m0 + ((lane & 16) ? (r + 8) : r);
      if (rr < n) {
        float v = acc[r] > 0.0f ? acc[r] : 0.0f;
        if (bias2) {
          v += bias2[j];
          v = v > 0.0f ? v : 0.0f;
        }
        out[(size_t)rr * NC + j] = v;
      }
    }
  }
}

extern "C" void kernel_launch(void* const* d_in, const int* in_sizes, int n_in,
                              void* d_out, int out_size, void* d_ws, size_t ws_size,
                              hipStream_t stream) {
  const float* x    = (const float*)d_in[0];
  const int*   esrc = (const int*)d_in[1];
  const int*   edst = (const int*)d_in[2];
  const float* W1   = (const float*)d_in[3];
  const float* b1   = (const float*)d_in[4];
  const float* W2   = (const float*)d_in[5];
  const float* b2   = (const float*)d_in[6];
  const float* eps  = (const float*)d_in[7];
  const float* bias = (const float*)d_in[8];

  const int E_ = in_sizes[1];
  const int n  = in_sizes[0] / NC;

  float*    h     = (float*)d_out;        // h buffer lives in d_out (in-place pipeline)
  _Float16* frags = (_Float16*)d_ws;      // 2 layers * 1024 frags * 16 halves = 64 KB

  // 1) h = (1+eps)*x
  int total4 = (n * NC) / 4;
  gin_init<<<(total4 + 255) / 256, 256, 0, stream>>>((const float4*)x, eps,
                                                     (float4*)h, total4);
  // 2) pack weights into WMMA fragment layout (tiny, one block)
  gin_pack_w<<<1, 256, 0, stream>>>(W1, W2, frags);

  // 3) h[dst] += x[src]  (hardware f32 atomics, fully coalesced row reads)
  long long items = (long long)E_ * 32;
  gin_scatter<<<(int)((items + 255) / 256), 256, 0, stream>>>(x, esrc, edst, h, E_);

  // 4) h = relu(h @ W1 + b1)   (in-place)
  int tiles  = (n + 15) / 16;
  int blocks = (tiles + 7) / 8;
  gin_gemm<<<blocks, 256, 0, stream>>>(h, h, frags, b1, nullptr, n);

  // 5) out = relu(relu(h @ W2 + b2) + bias)   (in-place, final output)
  gin_gemm<<<blocks, 256, 0, stream>>>(h, h, frags + (size_t)JT * KC * 32 * 16,
                                       b2, bias, n);
}
